// GNNStack_910533067064
// MI455X (gfx1250) — matrix operand order repaired and checked by reading, not compiled, not run on hardware
//
#include <hip/hip_runtime.h>

// ---------------------------------------------------------------------------
// CDNA5 (gfx1250, wave32) GNN stack. All GEMMs run through
// v_wmma_f32_16x16x32_bf16 with fp32 accumulators. Edge aggregation uses
// global fp32 atomics (L2). Layouts follow CDNA5 ISA tables.
// ---------------------------------------------------------------------------

typedef __attribute__((ext_vector_type(16))) __bf16 v16bf;
typedef __attribute__((ext_vector_type(8)))  __bf16 v8bf;
typedef __attribute__((ext_vector_type(8)))  float  v8f;

#define HID 256

// ---- pack fp32 weight [outF,K] row-major -> bf16 [Kpad,outF] (zero padded) ----
__global__ void k_pack_weight(const float* __restrict__ W, __bf16* __restrict__ Bt,
                              int outF, int K, int Kpad) {
  int idx = blockIdx.x * blockDim.x + threadIdx.x;
  int total = Kpad * outF;
  if (idx >= total) return;
  int k = idx / outF, n = idx - k * outF;
  float v = (k < K) ? W[(size_t)n * K + k] : 0.0f;
  Bt[idx] = (__bf16)v;
}

// ---- conv3x3 (8ch, 11x11 -> 9x9) + bias + concat bd + relu -> bf16 [N,672] ----
__global__ void k_conv_pack(const float* __restrict__ x, const float* __restrict__ bd,
                            const float* __restrict__ cw, const float* __restrict__ cb,
                            __bf16* __restrict__ h0, int N) {
  __shared__ float xs[8 * 11 * 11];
  __shared__ float ws[8 * 8 * 9];
  __shared__ float bs[8];
  int n = blockIdx.x, t = threadIdx.x;
  for (int i = t; i < 8 * 11 * 11; i += 256) xs[i] = x[(size_t)n * 968 + i];
  for (int i = t; i < 576; i += 256) ws[i] = cw[i];
  if (t < 8) bs[t] = cb[t];
  __syncthreads();
  for (int idx = t; idx < 672; idx += 256) {
    float v;
    if (idx < 648) {
      int co = idx / 81, r = idx - co * 81;
      int yy = r / 9, xx = r - yy * 9;
      float acc = bs[co];
#pragma unroll
      for (int ci = 0; ci < 8; ++ci)
#pragma unroll
        for (int dy = 0; dy < 3; ++dy)
#pragma unroll
          for (int dx = 0; dx < 3; ++dx)
            acc += xs[ci * 121 + (yy + dy) * 11 + (xx + dx)] *
                   ws[co * 72 + ci * 9 + dy * 3 + dx];
      v = fmaxf(acc, 0.0f);
    } else if (idx < 652) {
      v = fmaxf(bd[(size_t)n * 4 + (idx - 648)], 0.0f);
    } else {
      v = 0.0f;  // pad 652..671
    }
    h0[(size_t)n * 672 + idx] = (__bf16)v;
  }
}

// ---- WMMA GEMM: C[M,256] = A[M,Kpad](bf16) x Bt[Kpad,256](bf16) (+bias)(+=C) ----
// 256 threads = 8 wave32s; block tile 64x64; each wave computes 16x32.
__global__ void k_gemm_bf16(const __bf16* __restrict__ A, const __bf16* __restrict__ Bt,
                            const float* __restrict__ bias, float* __restrict__ C,
                            int M, int Kpad, int accumulate) {
  int lane = threadIdx.x & 31;
  int w = threadIdx.x >> 5;      // wave 0..7
  int wr = w & 3, wc = w >> 2;   // 4 row-groups x 2 col-groups
  int m0 = blockIdx.x * 64 + wr * 16;
  int n0 = blockIdx.y * 64 + wc * 32;
  int mrow = lane & 15, half = lane >> 4;
  int arow = m0 + mrow;
  int ar = (arow < M) ? arow : (M - 1);  // clamp; OOB rows never stored

  v8f c0 = {};
  v8f c1 = {};
  int nk = Kpad >> 5;
  for (int ks = 0; ks < nk; ++ks) {
    int k0 = ks << 5;
    // A fragment: lane (m, half) holds K = half*8+{0..7} and 16+half*8+{0..7}
    union { v16bf v; v8bf h[2]; } a;
    const __bf16* ap = A + (size_t)ar * Kpad + k0 + half * 8;
    a.h[0] = *(const v8bf*)ap;
    a.h[1] = *(const v8bf*)(ap + 16);
    // B fragment: lane (n, half) holds column n, K = half*8+{0..7} and 16+half*8+{0..7}
    const __bf16* bp = Bt + (size_t)k0 * HID + n0 + mrow;
    v16bf b0, b1;
#pragma unroll
    for (int s = 0; s < 16; ++s) {
      int kk = (s < 8) ? (half * 8 + s) : (16 + half * 8 + (s - 8));
      b0[s] = bp[(size_t)kk * HID];
      b1[s] = bp[(size_t)kk * HID + 16];
    }
    c0 = __builtin_amdgcn_wmma_f32_16x16x32_bf16(false, a.v, false, b0,
                                                 (short)0, c0, false, false);
    c1 = __builtin_amdgcn_wmma_f32_16x16x32_bf16(false, a.v, false, b1,
                                                 (short)0, c1, false, false);
  }
  // C/D layout: VGPR j, lane -> row m0 + j + 8*half, col n0 + (lane&15)
  int ccol = n0 + (lane & 15);
  int rbase = m0 + half * 8;
#pragma unroll
  for (int j = 0; j < 8; ++j) {
    int r = rbase + j;
    if (r < M) {
      float v0 = c0[j], v1 = c1[j];
      if (bias) { v0 += bias[ccol]; v1 += bias[ccol + 16]; }
      size_t i0 = (size_t)r * HID + ccol;
      if (accumulate) { C[i0] += v0; C[i0 + 16] += v1; }
      else            { C[i0]  = v0; C[i0 + 16]  = v1; }
    }
  }
}

// ---- fill fp32 ----
__global__ void k_fill(float* __restrict__ p, float v, long long n) {
  long long i = (long long)blockIdx.x * blockDim.x + threadIdx.x;
  if (i < n) p[i] = v;
}

// ---- scatter-add: agg[dst[e]] += feat[src[e]] over 256 features ----
__global__ void k_scatter_add(const float* __restrict__ feat, const int* __restrict__ src,
                              const int* __restrict__ dst, float* __restrict__ agg, int E) {
  for (int e = blockIdx.x; e < E; e += gridDim.x) {
    int s = src[e], d = dst[e];
    const float* f = feat + (size_t)s * HID;
    float* o = agg + (size_t)d * HID;
    for (int k = threadIdx.x; k < HID; k += blockDim.x)
      atomicAdd(o + k, f[k]);
  }
}

// ---- in-degree ----
__global__ void k_deg(const int* __restrict__ dst, float* __restrict__ deg, int E) {
  int i = blockIdx.x * blockDim.x + threadIdx.x;
  if (i < E) atomicAdd(&deg[dst[i]], 1.0f);
}

// ---- h = a + b, plus bf16 copy ----
__global__ void k_add_cvt(const float* __restrict__ a, const float* __restrict__ b,
                          float* __restrict__ h, __bf16* __restrict__ hbf, long long n) {
  long long i = (long long)blockIdx.x * blockDim.x + threadIdx.x;
  if (i < n) {
    float v = a[i] + b[i];
    h[i] = v;
    hbf[i] = (__bf16)v;
  }
}

// ---- mean aggregation: out_bf16 = agg / max(deg,1) ----
__global__ void k_meancvt(const float* __restrict__ agg, const float* __restrict__ deg,
                          __bf16* __restrict__ out, int N) {
  long long i = (long long)blockIdx.x * blockDim.x + threadIdx.x;
  if (i < (long long)N * HID) {
    int n = (int)(i >> 8);
    out[i] = (__bf16)(agg[i] / fmaxf(deg[n], 1.0f));
  }
}

// ---- per-node: optional emb copy, relu, optional layernorm, fp32+bf16 out ----
__global__ void k_post(const float* __restrict__ z, const float* __restrict__ g,
                       const float* __restrict__ b, float* __restrict__ h,
                       __bf16* __restrict__ hbf, float* __restrict__ emb, int do_ln) {
  __shared__ float red[HID];
  int n = blockIdx.x, t = threadIdx.x;
  size_t i = (size_t)n * HID + t;
  float v = z[i];
  if (emb) emb[i] = v;  // emb is pre-relu
  float r = fmaxf(v, 0.0f);
  float out = r;
  if (do_ln) {
    red[t] = r; __syncthreads();
    for (int s = 128; s > 0; s >>= 1) { if (t < s) red[t] += red[t + s]; __syncthreads(); }
    float mu = red[0] * (1.0f / HID); __syncthreads();
    float d = r - mu;
    red[t] = d * d; __syncthreads();
    for (int s = 128; s > 0; s >>= 1) { if (t < s) red[t] += red[t + s]; __syncthreads(); }
    float var = red[0] * (1.0f / HID);
    out = d * rsqrtf(var + 1e-5f) * g[t] + b[t];
  }
  h[i] = out;
  hbf[i] = (__bf16)out;
}

// ---- head: [256]->[5] projection + log_softmax (block per node) ----
__global__ void k_head(const float* __restrict__ z, const float* __restrict__ w2,
                       const float* __restrict__ b2, float* __restrict__ out) {
  __shared__ float zs[HID];
  __shared__ float dots[8];
  int n = blockIdx.x, t = threadIdx.x;
  zs[t] = z[(size_t)n * HID + t];
  __syncthreads();
  int o = t >> 5, ln = t & 31;   // wave32: one wave per output class
  if (o < 5) {
    float p = 0.0f;
    for (int k = ln; k < HID; k += 32) p += zs[k] * w2[o * HID + k];
    for (int m = 16; m >= 1; m >>= 1) p += __shfl_xor(p, m, 32);
    if (ln == 0) dots[o] = p + b2[o];
  }
  __syncthreads();
  if (t == 0) {
    float mx = dots[0];
    for (int q = 1; q < 5; ++q) mx = fmaxf(mx, dots[q]);
    float se = 0.0f;
    for (int q = 0; q < 5; ++q) se += expf(dots[q] - mx);
    float lse = mx + logf(se);
    for (int q = 0; q < 5; ++q) out[(size_t)n * 5 + q] = dots[q] - lse;
  }
}

extern "C" void kernel_launch(void* const* d_in, const int* in_sizes, int n_in,
                              void* d_out, int out_size, void* d_ws, size_t ws_size,
                              hipStream_t stream) {
  const float* x      = (const float*)d_in[0];
  const float* bd     = (const float*)d_in[1];
  const int*   ei     = (const int*)d_in[2];
  const float* cw     = (const float*)d_in[3];
  const float* cb     = (const float*)d_in[4];
  const float* lin_w  = (const float*)d_in[5];
  const float* lin_b  = (const float*)d_in[6];
  const float* lins_w = (const float*)d_in[7];
  const float* lins_b = (const float*)d_in[8];
  const float* sl_w   = (const float*)d_in[9];
  const float* sl_b   = (const float*)d_in[10];
  const float* sr_w   = (const float*)d_in[11];
  const float* lng    = (const float*)d_in[12];
  const float* lnb    = (const float*)d_in[13];
  const float* mp1_w  = (const float*)d_in[14];
  const float* mp1_b  = (const float*)d_in[15];
  const float* mp2_w  = (const float*)d_in[16];
  const float* mp2_b  = (const float*)d_in[17];

  int N = in_sizes[1] / 4;   // bd_pred is [N,4]
  int E = in_sizes[2] / 2;   // edge_index is [2,E]
  const int K0 = 652, K0P = 672, H = HID;

  // workspace sub-allocation (256B aligned)
  char* p = (char*)d_ws;
  auto alloc = [&](size_t bytes) {
    void* r = (void*)p;
    p += (bytes + 255) & ~(size_t)255;
    return r;
  };
  __bf16* h0bf    = (__bf16*)alloc((size_t)N * K0P * 2);
  __bf16* wt_lin  = (__bf16*)alloc((size_t)K0P * H * 2);
  __bf16* wt_lins = (__bf16*)alloc((size_t)K0P * H * 2);
  __bf16* wt_sl   = (__bf16*)alloc((size_t)3 * H * H * 2);
  __bf16* wt_sr   = (__bf16*)alloc((size_t)3 * H * H * 2);
  __bf16* wt_mp1  = (__bf16*)alloc((size_t)H * H * 2);
  float*  buf0    = (float*)alloc((size_t)N * H * 4);   // self_x, later z (head)
  float*  buf1    = (float*)alloc((size_t)N * H * 4);   // neigh, later sage z
  float*  agg     = (float*)alloc((size_t)N * H * 4);
  float*  hbuf    = (float*)alloc((size_t)N * H * 4);
  __bf16* hbf     = (__bf16*)alloc((size_t)N * H * 2);
  __bf16* aggbf   = (__bf16*)alloc((size_t)N * H * 2);
  float*  deg     = (float*)alloc((size_t)N * 4);

  const int* src = ei;
  const int* dst = ei + E;

  // ---- pack weights to bf16, transposed [Kpad, 256] ----
  int tpW = (K0P * H + 255) / 256;
  int tpH = (H * H + 255) / 256;
  k_pack_weight<<<tpW, 256, 0, stream>>>(lin_w,  wt_lin,  H, K0, K0P);
  k_pack_weight<<<tpW, 256, 0, stream>>>(lins_w, wt_lins, H, K0, K0P);
  for (int i = 0; i < 3; ++i) {
    k_pack_weight<<<tpH, 256, 0, stream>>>(sl_w + (size_t)i * H * H, wt_sl + (size_t)i * H * H, H, H, H);
    k_pack_weight<<<tpH, 256, 0, stream>>>(sr_w + (size_t)i * H * H, wt_sr + (size_t)i * H * H, H, H, H);
  }
  k_pack_weight<<<tpH, 256, 0, stream>>>(mp1_w, wt_mp1, H, H, H);

  // ---- conv + concat + relu -> bf16 features ----
  k_conv_pack<<<N, 256, 0, stream>>>(x, bd, cw, cb, h0bf, N);

  // ---- layer 0: two WMMA GEMMs, scatter-add, h = self + agg ----
  dim3 gg((N + 63) / 64, H / 64);
  k_gemm_bf16<<<gg, 256, 0, stream>>>(h0bf, wt_lins, lins_b, buf0, N, K0P, 0);
  k_gemm_bf16<<<gg, 256, 0, stream>>>(h0bf, wt_lin,  lin_b,  buf1, N, K0P, 0);

  long long nh = (long long)N * H;
  int gnh = (int)((nh + 255) / 256);
  int sg = E < 131072 ? E : 131072;
  k_fill<<<gnh, 256, 0, stream>>>(agg, 0.0f, nh);
  k_fill<<<(N + 255) / 256, 256, 0, stream>>>(deg, 0.0f, (long long)N);
  k_scatter_add<<<sg, 256, 0, stream>>>(buf1, src, dst, agg, E);
  k_deg<<<(E + 255) / 256, 256, 0, stream>>>(dst, deg, E);
  k_add_cvt<<<gnh, 256, 0, stream>>>(buf0, agg, hbuf, hbf, nh);

  // ---- SAGE layers 1..3 ----
  for (int i = 0; i < 3; ++i) {
    k_fill<<<gnh, 256, 0, stream>>>(agg, 0.0f, nh);
    k_scatter_add<<<sg, 256, 0, stream>>>(hbuf, src, dst, agg, E);
    k_meancvt<<<gnh, 256, 0, stream>>>(agg, deg, aggbf, N);
    k_gemm_bf16<<<gg, 256, 0, stream>>>(aggbf, wt_sl + (size_t)i * H * H, sl_b + (size_t)i * H,
                                        buf1, N, H, 0);
    k_gemm_bf16<<<gg, 256, 0, stream>>>(hbf, wt_sr + (size_t)i * H * H, nullptr,
                                        buf1, N, H, 1);
    int li = (i < 2) ? i : 0;  // g/b only dereferenced when do_ln=1
    float* emb = (i == 2) ? (float*)d_out : nullptr;
    k_post<<<N, 256, 0, stream>>>(buf1, lng + (size_t)li * H, lnb + (size_t)li * H,
                                  hbuf, hbf, emb, (i < 2) ? 1 : 0);
  }

  // ---- head: mp1 GEMM (WMMA) + mp2 projection + log_softmax ----
  k_gemm_bf16<<<gg, 256, 0, stream>>>(hbf, wt_mp1, mp1_b, buf0, N, H, 0);
  k_head<<<N, 256, 0, stream>>>(buf0, mp2_w, mp2_b, (float*)d_out + (size_t)N * H);

  (void)n_in; (void)out_size; (void)ws_size;
}
